// DFlashDraftModel_62122406969391
// MI455X (gfx1250) — matrix operand order repaired and checked by reading, not compiled, not run on hardware
//
#include <hip/hip_runtime.h>
#include <hip/hip_bf16.h>
#include <math.h>

// ---- model constants (from reference) ----
#define L_    5
#define B_    4
#define Q_    16
#define CTX_  2048
#define S_    2064        // CTX + Q
#define SP_   2080        // S padded to multiple of 32 for WMMA K-loop
#define H_    2560
#define I_    6912
#define NH_   20
#define NKV_  4
#define HD_   128
#define NF_   5
#define GQA_  5           // NH / NKV
#define EPS_  1e-6f
#define LOG_THETA_ 13.815510557964274f  // ln(1e6)

#define WROW_ 40          // LDS row stride (floats): 160B -> b128-aligned, spreads banks

typedef __attribute__((ext_vector_type(16))) __bf16 v16bf;
typedef __attribute__((ext_vector_type(8)))  float  v8f;

// CDNA5 async Global->LDS path (ASYNCcnt-tracked).  The clang builtin on this
// toolchain has a buffer-rsrc-style prototype, so we drive the instruction via
// inline asm (VGLOBAL encoding: VDST = LDS byte address, VADDR = 64-bit global
// address; see CDNA5 ISA).
#if defined(__gfx1250__)
#define USE_ASYNC_LDS 1
#else
#define USE_ASYNC_LDS 0
#endif

#if USE_ASYNC_LDS
typedef __attribute__((address_space(3))) void lds_void_t;
__device__ __forceinline__ void async_copy16(const float* g, float* l) {
  unsigned lds_off = (unsigned)(size_t)(lds_void_t*)l;   // AS3 ptr == LDS byte offset
  asm volatile("global_load_async_to_lds_b128 %0, %1, off"
               :
               : "v"(lds_off), "v"((unsigned long long)(size_t)g)
               : "memory");
}
__device__ __forceinline__ void wait_async_le4() {
  asm volatile("s_wait_asynccnt 4" ::: "memory");
}
__device__ __forceinline__ void wait_async_0() {
  asm volatile("s_wait_asynccnt 0" ::: "memory");
}
#endif

// ---------------------------------------------------------------------------
// bf16 fragment loaders (fp32 -> bf16 in-register; v_cvt co-issues with the
// WMMA/XDL pipe).  Lane layout used here (contiguous k, not AMD's table order):
//  A (16x32, 16-bit): lanes 0-15 = rows, K 0..15;
//                     lanes 16-31 = rows, K 16..31.
//  B (32x16, 16-bit): lanes 0-15 = cols, K 0..15; lanes 16-31 = cols, K 16..31.
// ---------------------------------------------------------------------------
__device__ __forceinline__ v16bf frag_a_f32(const float* __restrict__ p, int sel) {
  v16bf r;
  const float* p0 = p + sel * 16;       // K group 0
  const float* p1 = p + sel * 16 + 8;   // K group 1
#pragma unroll
  for (int j = 0; j < 8; ++j) { r[j] = (__bf16)p0[j]; r[j + 8] = (__bf16)p1[j]; }
  return r;
}
__device__ __forceinline__ v16bf frag_b_f32(const float* __restrict__ p) {
  v16bf r;
#pragma unroll
  for (int j = 0; j < 16; ++j) r[j] = (__bf16)p[j];
  return r;
}

// ---------------------------------------------------------------------------
// Generic NT GEMM:  C[M,N] = A[M,K] * W[N,K]^T (+ resid), fp32 I/O, bf16 WMMA.
// Block = 8 waves, 64(M) x 128(N); each wave: 16 rows x 64 cols (4 accs).
// Weight tile (128 x 32 fp32) is double-buffered in LDS via async Global->LDS
// copies, software-pipelined one K-stage ahead.  Grids always tile M,N exactly.
// ---------------------------------------------------------------------------
__global__ __launch_bounds__(256) void gemm_nt_bf16(
    const float* __restrict__ A, const float* __restrict__ W,
    const float* __restrict__ resid, float* __restrict__ C,
    int M, int N, int K, int lda, int ldw, int ldc)
{
  __shared__ float lds_w[2][128 * WROW_];       // 2 x 20 KB
  const int tid  = threadIdx.x;
  const int lane = tid & 31;
  const int wave = tid >> 5;
  const int wm = wave >> 1, wn = wave & 1;
  const int m0 = blockIdx.y * 64 + wm * 16;
  const int sel = lane >> 4;
  const int l15 = lane & 15;

  const float* wblk = W + (size_t)(blockIdx.x * 128) * ldw;   // 128 weight rows
  const float* arow = A + (size_t)(m0 + l15) * lda;

  // stage copy: 1024 x 16B chunks, 4 per thread
  auto issue_stage = [&](int kk, int buf) {
#pragma unroll
    for (int c = 0; c < 4; ++c) {
      int chunk = tid + c * 256;                // 0..1023
      int row = chunk >> 3, col = chunk & 7;    // 8 chunks per 32-float row
      const float* g = wblk + (size_t)row * ldw + kk + col * 4;
      float* l = &lds_w[buf][row * WROW_ + col * 4];
#if USE_ASYNC_LDS
      async_copy16(g, l);
#else
      *(float4*)l = *(const float4*)g;
#endif
    }
  };

  v8f acc0 = {}, acc1 = {}, acc2 = {}, acc3 = {};
  const int nsteps = K / 32;
  issue_stage(0, 0);
  for (int s = 0; s < nsteps; ++s) {
    const int kk = s * 32;
    const bool hasnext = (s + 1 < nsteps);
    if (hasnext) issue_stage(kk + 32, (s + 1) & 1);
#if USE_ASYNC_LDS
    // per-wave in-order completion: retire current stage (4 ops), keep next in flight
    if (hasnext) wait_async_le4();
    else         wait_async_0();
#endif
    __syncthreads();                            // tile visible to all waves
    __builtin_prefetch(arow + kk + 256, 0, 1);  // global_prefetch_b8 on A stream
    v16bf a = frag_a_f32(arow + kk, sel);
    const float* lw = &lds_w[s & 1][(wn * 64 + l15) * WROW_ + sel * 16];
    v16bf b0 = frag_b_f32(lw);
    v16bf b1 = frag_b_f32(lw + 16 * WROW_);
    v16bf b2 = frag_b_f32(lw + 32 * WROW_);
    v16bf b3 = frag_b_f32(lw + 48 * WROW_);
    acc0 = __builtin_amdgcn_wmma_f32_16x16x32_bf16(false, a, false, b0, (short)0, acc0, false, false);
    acc1 = __builtin_amdgcn_wmma_f32_16x16x32_bf16(false, a, false, b1, (short)0, acc1, false, false);
    acc2 = __builtin_amdgcn_wmma_f32_16x16x32_bf16(false, a, false, b2, (short)0, acc2, false, false);
    acc3 = __builtin_amdgcn_wmma_f32_16x16x32_bf16(false, a, false, b3, (short)0, acc3, false, false);
    __syncthreads();                            // done reading before overwrite
  }

  // C/D layout: VGPR j holds row (m0 + sel*8 + j), col (n0 + l15 + t*16)
  const int crow0 = m0 + sel * 8;
  const int ccol  = blockIdx.x * 128 + wn * 64 + l15;
#pragma unroll
  for (int j = 0; j < 8; ++j) {
    size_t r = (size_t)(crow0 + j) * ldc + ccol;
    float v0 = acc0[j], v1 = acc1[j], v2 = acc2[j], v3 = acc3[j];
    if (resid) { v0 += resid[r]; v1 += resid[r + 16]; v2 += resid[r + 32]; v3 += resid[r + 48]; }
    C[r] = v0; C[r + 16] = v1; C[r + 32] = v2; C[r + 48] = v3;
  }
}

// ---------------------------------------------------------------------------
// Attention scores: one wave per (b,h,16-key tile); 16x16 tile, K = HD = 128.
// ---------------------------------------------------------------------------
__global__ __launch_bounds__(32) void attn_scores(
    const float* __restrict__ QH, const float* __restrict__ KH, float* __restrict__ SC)
{
  const int lane = threadIdx.x & 31;
  const int sel = lane >> 4, l15 = lane & 15;
  const int st = blockIdx.x * 16;          // key tile
  const int bh = blockIdx.y;
  const int b = bh / NH_, h = bh % NH_;
  const int kv = h / GQA_;
  const float* qrow = QH + (size_t)(b * Q_ + l15) * (NH_ * HD_) + h * HD_;
  const float* krow = KH + ((size_t)(b * NKV_ + kv) * SP_ + st + l15) * HD_ + sel * 16;
  v8f acc = {};
#pragma unroll
  for (int kk = 0; kk < HD_; kk += 32) {
    v16bf a = frag_a_f32(qrow + kk, sel);
    v16bf bf = frag_b_f32(krow + kk);
    acc = __builtin_amdgcn_wmma_f32_16x16x32_bf16(false, a, false, bf, (short)0, acc, false, false);
  }
#pragma unroll
  for (int j = 0; j < 8; ++j)
    SC[(size_t)(bh * Q_ + sel * 8 + j) * SP_ + st + l15] = acc[j];
}

// ---------------------------------------------------------------------------
// Attention P*V: one block per (b,h); 8 waves each own a 16-wide slice of HD.
// K dim = SP_ (pad probs are 0, pad V rows are 0).
// ---------------------------------------------------------------------------
__global__ __launch_bounds__(256) void attn_pv(
    const float* __restrict__ SC, const float* __restrict__ VH, float* __restrict__ O)
{
  const int lane = threadIdx.x & 31;
  const int wave = threadIdx.x >> 5;          // n-subtile 0..7
  const int sel = lane >> 4, l15 = lane & 15;
  const int bh = blockIdx.x;
  const int b = bh / NH_, h = bh % NH_;
  const int kv = h / GQA_;
  const float* prow  = SC + (size_t)(bh * Q_ + l15) * SP_;
  const float* vbase = VH + (size_t)(b * NKV_ + kv) * SP_ * HD_ + wave * 16 + l15;
  v8f acc = {};
  for (int kk = 0; kk < SP_; kk += 32) {
    v16bf a = frag_a_f32(prow + kk, sel);
    const float* vp = vbase + (size_t)(kk + sel * 16) * HD_;
    v16bf bf;
#pragma unroll
    for (int j = 0; j < 16; ++j) bf[j] = (__bf16)vp[(size_t)j * HD_];
    acc = __builtin_amdgcn_wmma_f32_16x16x32_bf16(false, a, false, bf, (short)0, acc, false, false);
  }
#pragma unroll
  for (int j = 0; j < 8; ++j)
    O[(size_t)(b * Q_ + sel * 8 + j) * (NH_ * HD_) + h * HD_ + wave * 16 + l15] = acc[j];
}

// ---------------------------------------------------------------------------
// RMSNorm over contiguous rows (len = H or HD).
// ---------------------------------------------------------------------------
__global__ __launch_bounds__(256) void rmsnorm_rows(
    const float* __restrict__ X, const float* __restrict__ W,
    float* __restrict__ Y, int len)
{
  __shared__ float red[256];
  const float* x = X + (size_t)blockIdx.x * len;
  float s = 0.f;
  for (int i = threadIdx.x; i < len; i += 256) { float v = x[i]; s += v * v; }
  red[threadIdx.x] = s; __syncthreads();
  for (int off = 128; off > 0; off >>= 1) {
    if ((int)threadIdx.x < off) red[threadIdx.x] += red[threadIdx.x + off];
    __syncthreads();
  }
  float inv = rsqrtf(red[0] / (float)len + EPS_);
  float* y = Y + (size_t)blockIdx.x * len;
  for (int i = threadIdx.x; i < len; i += 256) y[i] = W[i] * x[i] * inv;
}

// ---------------------------------------------------------------------------
// Softmax over rows of SC (1/sqrt(HD) folded in); zero pad columns [S_, SP_).
// ---------------------------------------------------------------------------
__global__ __launch_bounds__(256) void softmax_row(float* __restrict__ SC)
{
  __shared__ float red[256];
  float* x = SC + (size_t)blockIdx.x * SP_;
  const float scale = 0.08838834764831845f;
  float m = -1e30f;
  for (int i = threadIdx.x; i < S_; i += 256) m = fmaxf(m, x[i]);
  red[threadIdx.x] = m; __syncthreads();
  for (int off = 128; off > 0; off >>= 1) {
    if ((int)threadIdx.x < off) red[threadIdx.x] = fmaxf(red[threadIdx.x], red[threadIdx.x + off]);
    __syncthreads();
  }
  m = red[0]; __syncthreads();
  float s = 0.f;
  for (int i = threadIdx.x; i < S_; i += 256) {
    float e = expf((x[i] - m) * scale); x[i] = e; s += e;
  }
  red[threadIdx.x] = s; __syncthreads();
  for (int off = 128; off > 0; off >>= 1) {
    if ((int)threadIdx.x < off) red[threadIdx.x] += red[threadIdx.x + off];
    __syncthreads();
  }
  float inv = 1.f / red[0];
  for (int i = threadIdx.x; i < SP_; i += 256) x[i] = (i < S_) ? x[i] * inv : 0.f;
}

// ---------------------------------------------------------------------------
// RoPE helpers / gather kernels
// ---------------------------------------------------------------------------
__device__ __forceinline__ void rope_pair(float& lo, float& hi, float pos, int dh) {
  float inv = expf(-((float)(2 * dh) / (float)HD_) * LOG_THETA_);
  float a = pos * inv;
  float c = cosf(a), s = sinf(a);
  float nlo = lo * c - hi * s;
  float nhi = hi * c + lo * s;
  lo = nlo; hi = nhi;
}

__global__ void rope_q(float* __restrict__ QH, const int* __restrict__ pos_ids)
{
  int idx = blockIdx.x * blockDim.x + threadIdx.x;       // B*Q*NH*(HD/2)
  if (idx >= B_ * Q_ * NH_ * (HD_ / 2)) return;
  int dh = idx & 63; int t = idx >> 6;
  int h = t % NH_; t /= NH_;
  int q = t % Q_;  int b = t / Q_;
  float pos = (float)pos_ids[b * S_ + CTX_ + q];
  float* p = QH + (size_t)(b * Q_ + q) * (NH_ * HD_) + h * HD_ + dh;
  float lo = p[0], hi = p[64];
  rope_pair(lo, hi, pos, dh);
  p[0] = lo; p[64] = hi;
}

__global__ void build_k(const float* __restrict__ KC, const float* __restrict__ KD,
                        const int* __restrict__ pos_ids, float* __restrict__ KH)
{
  int idx = blockIdx.x * blockDim.x + threadIdx.x;       // B*NKV*SP*(HD/2)
  if (idx >= B_ * NKV_ * SP_ * (HD_ / 2)) return;
  int dh = idx & 63; int t = idx >> 6;
  int s = t % SP_; t /= SP_;
  int kv = t % NKV_; int b = t / NKV_;
  float* dst = KH + ((size_t)(b * NKV_ + kv) * SP_ + s) * HD_ + dh;
  if (s >= S_) { dst[0] = 0.f; dst[64] = 0.f; return; }
  float lo, hi;
  if (s < CTX_) {
    const float* src = KC + (size_t)(b * CTX_ + s) * (NKV_ * HD_) + kv * HD_ + dh;
    lo = src[0]; hi = src[64];
  } else {
    const float* src = KD + (size_t)(b * Q_ + (s - CTX_)) * (NKV_ * HD_) + kv * HD_ + dh;
    lo = src[0]; hi = src[64];
  }
  rope_pair(lo, hi, (float)pos_ids[b * S_ + s], dh);
  dst[0] = lo; dst[64] = hi;
}

__global__ void build_v(const float* __restrict__ VC, const float* __restrict__ VD,
                        float* __restrict__ VH)
{
  int idx = blockIdx.x * blockDim.x + threadIdx.x;       // B*NKV*SP*HD
  if (idx >= B_ * NKV_ * SP_ * HD_) return;
  int d = idx & (HD_ - 1); int t = idx >> 7;
  int s = t % SP_; t /= SP_;
  int kv = t % NKV_; int b = t / NKV_;
  float v;
  if (s >= S_)        v = 0.f;
  else if (s < CTX_)  v = VC[(size_t)(b * CTX_ + s) * (NKV_ * HD_) + kv * HD_ + d];
  else                v = VD[(size_t)(b * Q_ + (s - CTX_)) * (NKV_ * HD_) + kv * HD_ + d];
  VH[((size_t)(b * NKV_ + kv) * SP_ + s) * HD_ + d] = v;
}

__global__ void silu_mul(const float* __restrict__ G, const float* __restrict__ U,
                         float* __restrict__ GU, int n)
{
  int i = blockIdx.x * blockDim.x + threadIdx.x;
  if (i >= n) return;
  float g = G[i];
  GU[i] = (g / (1.f + expf(-g))) * U[i];
}

__global__ void copy_f32(const float* __restrict__ src, float* __restrict__ dst, int n)
{
  int i = blockIdx.x * blockDim.x + threadIdx.x;
  if (i < n) dst[i] = src[i];
}

// ---------------------------------------------------------------------------
extern "C" void kernel_launch(void* const* d_in, const int* in_sizes, int n_in,
                              void* d_out, int out_size, void* d_ws, size_t ws_size,
                              hipStream_t stream)
{
  (void)in_sizes; (void)n_in; (void)out_size; (void)ws_size;
  const float* noise   = (const float*)d_in[0];
  const float* target  = (const float*)d_in[1];
  const int*   pos     = (const int*)  d_in[2];
  const float* fc_w    = (const float*)d_in[3];
  const float* hn_w    = (const float*)d_in[4];
  const float* q_w     = (const float*)d_in[5];
  const float* k_w     = (const float*)d_in[6];
  const float* v_w     = (const float*)d_in[7];
  const float* o_w     = (const float*)d_in[8];
  const float* qn_w    = (const float*)d_in[9];
  const float* kn_w    = (const float*)d_in[10];
  const float* gate_w  = (const float*)d_in[11];
  const float* up_w    = (const float*)d_in[12];
  const float* down_w  = (const float*)d_in[13];
  const float* ln1_w   = (const float*)d_in[14];
  const float* ln2_w   = (const float*)d_in[15];
  const float* norm_w  = (const float*)d_in[16];
  float* out = (float*)d_out;

  // workspace carve-out (~171 MB of fp32)
  float* w = (float*)d_ws;
  size_t off = 0;
  auto alloc = [&](size_t n) { float* p = w + off; off += n; return p; };
  const int MQ = B_ * Q_;                 // 64 decode rows
  const int MC = B_ * CTX_;               // 8192 context rows
  float* TH = alloc((size_t)MC * H_);
  float* KC = alloc((size_t)MC * NKV_ * HD_);
  float* VC = alloc((size_t)MC * NKV_ * HD_);
  float* KD = alloc((size_t)MQ * NKV_ * HD_);
  float* VD = alloc((size_t)MQ * NKV_ * HD_);
  float* QH = alloc((size_t)MQ * NH_ * HD_);
  float* KH = alloc((size_t)B_ * NKV_ * SP_ * HD_);
  float* VH = alloc((size_t)B_ * NKV_ * SP_ * HD_);
  float* SC = alloc((size_t)B_ * NH_ * Q_ * SP_);
  float* O  = alloc((size_t)MQ * NH_ * HD_);
  float* X  = alloc((size_t)MQ * H_);
  float* HN = alloc((size_t)MQ * H_);
  float* H2 = alloc((size_t)MQ * H_);
  float* G  = alloc((size_t)MQ * I_);
  float* U  = alloc((size_t)MQ * I_);
  float* GU = alloc((size_t)MQ * I_);

  const int nX = MQ * H_;
  copy_f32<<<(nX + 255) / 256, 256, 0, stream>>>(noise, X, nX);

  // th = rms(target_hidden @ fc_w^T, hidden_norm_w)
  gemm_nt_bf16<<<dim3(H_ / 128, MC / 64), 256, 0, stream>>>(
      target, fc_w, nullptr, TH, MC, H_, NF_ * H_, NF_ * H_, NF_ * H_, H_);
  rmsnorm_rows<<<MC, 256, 0, stream>>>(TH, hn_w, TH, H_);

  for (int l = 0; l < L_; ++l) {
    const float* qw = q_w    + (size_t)l * NH_  * HD_ * H_;
    const float* kw = k_w    + (size_t)l * NKV_ * HD_ * H_;
    const float* vw = v_w    + (size_t)l * NKV_ * HD_ * H_;
    const float* ow = o_w    + (size_t)l * H_ * (NH_ * HD_);
    const float* gw = gate_w + (size_t)l * I_ * H_;
    const float* uw = up_w   + (size_t)l * I_ * H_;
    const float* dw = down_w + (size_t)l * H_ * I_;

    rmsnorm_rows<<<MQ, 256, 0, stream>>>(X, ln1_w + (size_t)l * H_, HN, H_);

    gemm_nt_bf16<<<dim3((NH_ * HD_) / 128, MQ / 64), 256, 0, stream>>>(
        HN, qw, nullptr, QH, MQ, NH_ * HD_, H_, H_, H_, NH_ * HD_);
    gemm_nt_bf16<<<dim3((NKV_ * HD_) / 128, MQ / 64), 256, 0, stream>>>(
        HN, kw, nullptr, KD, MQ, NKV_ * HD_, H_, H_, H_, NKV_ * HD_);
    gemm_nt_bf16<<<dim3((NKV_ * HD_) / 128, MQ / 64), 256, 0, stream>>>(
        HN, vw, nullptr, VD, MQ, NKV_ * HD_, H_, H_, H_, NKV_ * HD_);
    gemm_nt_bf16<<<dim3((NKV_ * HD_) / 128, MC / 64), 256, 0, stream>>>(
        TH, kw, nullptr, KC, MC, NKV_ * HD_, H_, H_, H_, NKV_ * HD_);
    gemm_nt_bf16<<<dim3((NKV_ * HD_) / 128, MC / 64), 256, 0, stream>>>(
        TH, vw, nullptr, VC, MC, NKV_ * HD_, H_, H_, H_, NKV_ * HD_);

    // per-head RMS of q/k (rows of HD are contiguous in these layouts)
    rmsnorm_rows<<<MQ * NH_, 256, 0, stream>>>(QH, qn_w + (size_t)l * HD_, QH, HD_);
    rmsnorm_rows<<<MC * NKV_, 256, 0, stream>>>(KC, kn_w + (size_t)l * HD_, KC, HD_);
    rmsnorm_rows<<<MQ * NKV_, 256, 0, stream>>>(KD, kn_w + (size_t)l * HD_, KD, HD_);

    int nq = B_ * Q_ * NH_ * (HD_ / 2);
    rope_q<<<(nq + 255) / 256, 256, 0, stream>>>(QH, pos);
    int nk = B_ * NKV_ * SP_ * (HD_ / 2);
    build_k<<<(nk + 255) / 256, 256, 0, stream>>>(KC, KD, pos, KH);
    int nv = B_ * NKV_ * SP_ * HD_;
    build_v<<<(nv + 255) / 256, 256, 0, stream>>>(VC, VD, VH);

    attn_scores<<<dim3(S_ / 16, B_ * NH_), 32, 0, stream>>>(QH, KH, SC);
    softmax_row<<<B_ * NH_ * Q_, 256, 0, stream>>>(SC);
    attn_pv<<<B_ * NH_, 256, 0, stream>>>(SC, VH, O);

    // x += O @ o_w^T
    gemm_nt_bf16<<<dim3(H_ / 128, MQ / 64), 256, 0, stream>>>(
        O, ow, X, X, MQ, H_, NH_ * HD_, NH_ * HD_, NH_ * HD_, H_);

    rmsnorm_rows<<<MQ, 256, 0, stream>>>(X, ln2_w + (size_t)l * H_, H2, H_);
    gemm_nt_bf16<<<dim3(I_ / 128, MQ / 64), 256, 0, stream>>>(
        H2, gw, nullptr, G, MQ, I_, H_, H_, H_, I_);
    gemm_nt_bf16<<<dim3(I_ / 128, MQ / 64), 256, 0, stream>>>(
        H2, uw, nullptr, U, MQ, I_, H_, H_, H_, I_);
    int ni = MQ * I_;
    silu_mul<<<(ni + 255) / 256, 256, 0, stream>>>(G, U, GU, ni);
    gemm_nt_bf16<<<dim3(H_ / 128, MQ / 64), 256, 0, stream>>>(
        GU, dw, X, X, MQ, H_, I_, I_, I_, H_);
  }

  rmsnorm_rows<<<MQ, 256, 0, stream>>>(X, norm_w, out, H_);
}
